// GAT_9294309228706
// MI455X (gfx1250) — compile-verified, matrix operand order checked by reading, not run on hardware
//
#include <hip/hip_runtime.h>

typedef __attribute__((ext_vector_type(16))) __bf16 v16bf;
typedef __attribute__((ext_vector_type(8)))  float  v8f;

#define HIDC   256
#define HEADS  8
#define WCOLS  2048   // HEADS * 256
#define DOCIN  3072
#define LDSP   264    // 256 + 8 bf16 pad -> row stride 528B, conflict-free frag reads

union FragBF { v16bf v; uint4 q[2]; };

// ---------------------------------------------------------------- conversions
__global__ __launch_bounds__(256) void k_f32_to_bf16(const float* __restrict__ in,
                                                     __bf16* __restrict__ out, size_t n) {
    size_t i = (size_t)blockIdx.x * 256 + threadIdx.x;
    if (i < n) out[i] = (__bf16)in[i];
}

// W [256, 2048] f32 row-major  ->  Wt [2048, 256] bf16 (N-major, i.e. B^T)
__global__ __launch_bounds__(256) void k_transpose_w(const float* __restrict__ W,
                                                     __bf16* __restrict__ Wt) {
    int i = blockIdx.x * 256 + threadIdx.x;   // i = r*256 + c  (r = out col, c = k)
    int r = i >> 8, c = i & 255;
    Wt[i] = (__bf16)W[(size_t)c * WCOLS + r];
}

// ---------------------------------------------------------------- WMMA GEMM
// C[M, 2048] f32 = A[M, 256] bf16 @ Wt[2048, 256]^T
// Block: 256 thr = 8 waves; block tile 32(M) x 256(N); wave tile 16 x 64.
// A tile staged into LDS via CDNA5 async-to-LDS engine (ASYNCcnt).
__global__ __launch_bounds__(256) void k_gemm_wmma(const __bf16* __restrict__ A,
                                                   const __bf16* __restrict__ Bt,
                                                   float* __restrict__ C, int M) {
    __shared__ __bf16 lds_a[32 * LDSP];
    const int tid  = threadIdx.x;
    const int wave = tid >> 5;
    const int lane = tid & 31;
    const int row0 = blockIdx.x * 32;
    const int wrow = (wave >> 2) << 4;                       // 0 or 16
    const int wcol = blockIdx.y * 256 + (wave & 3) * 64;

    // cooperative A tile load: 32 rows x 256 bf16, 16B per thread x4,
    // async global -> LDS (no VGPR round-trip), tracked with ASYNCcnt
    #pragma unroll
    for (int i = 0; i < 4; ++i) {
        int chunk = tid + (i << 8);          // 0..1023
        int r  = chunk >> 5;
        int ck = (chunk & 31) << 3;
        int gr = row0 + r; if (gr >= M) gr = M - 1;
        unsigned long long ga = (unsigned long long)(A + (size_t)gr * HIDC + ck);
        unsigned int       la = (unsigned int)(uintptr_t)(&lds_a[r * LDSP + ck]);
        asm volatile("global_load_async_to_lds_b128 %0, %1, off"
                     :: "v"(la), "v"(ga) : "memory");
    }
    asm volatile("s_wait_asynccnt 0" ::: "memory");
    __syncthreads();

    v8f acc[4] = {};
    const int halfk = (lane < 16) ? 0 : 8;   // ISA A/B frag: hi half-wave holds K+8 / K+24
    const int arow  = wrow + (lane & 15);

    for (int k0 = 0; k0 < HIDC; k0 += 32) {
        FragBF a;
        const __bf16* ap = &lds_a[arow * LDSP + k0 + halfk];
        a.q[0] = *reinterpret_cast<const uint4*>(ap);
        a.q[1] = *reinterpret_cast<const uint4*>(ap + 16);
        #pragma unroll
        for (int nt = 0; nt < 4; ++nt) {
            const __bf16* bp =
                Bt + (size_t)(wcol + (nt << 4) + (lane & 15)) * HIDC + k0 + halfk;
            if (k0 + 32 < HIDC) __builtin_prefetch(bp + 32, 0, 1); // global_prefetch_b8
            FragBF b;
            b.q[0] = *reinterpret_cast<const uint4*>(bp);
            b.q[1] = *reinterpret_cast<const uint4*>(bp + 16);
            acc[nt] = __builtin_amdgcn_wmma_f32_16x16x32_bf16(
                false, a.v, false, b.v, (short)0, acc[nt], false, false);
        }
    }

    const int rbase = row0 + wrow + ((lane < 16) ? 0 : 8);
    #pragma unroll
    for (int nt = 0; nt < 4; ++nt) {
        int cc = wcol + (nt << 4) + (lane & 15);
        #pragma unroll
        for (int v = 0; v < 8; ++v) {
            int r = rbase + v;
            if (r < M) C[(size_t)r * WCOLS + cc] = acc[nt][v];
        }
    }
}

// ---------------------------------------------------------------- attention logits
// one wave32 per (node, head): 256-length dual dot product
__global__ __launch_bounds__(256) void k_attn_logits(const float* __restrict__ h,
        const float* __restrict__ a_s, const float* __restrict__ a_d,
        float* __restrict__ al_s, float* __restrict__ al_d, int N) {
    int warp = (blockIdx.x * 256 + threadIdx.x) >> 5;
    int lane = threadIdx.x & 31;
    if (warp >= N * HEADS) return;
    int n = warp >> 3, hh = warp & 7;
    const float* hp = h + (size_t)n * WCOLS + hh * 256;
    const float* as = a_s + hh * 256;
    const float* ad = a_d + hh * 256;
    float s = 0.f, d = 0.f;
    for (int k = lane; k < 256; k += 32) { float v = hp[k]; s += v * as[k]; d += v * ad[k]; }
    #pragma unroll
    for (int off = 16; off; off >>= 1) {
        s += __shfl_down(s, off, 32);
        d += __shfl_down(d, off, 32);
    }
    if (lane == 0) { al_s[warp] = s; al_d[warp] = d; }
}

// ---------------------------------------------------------------- edge softmax
__global__ __launch_bounds__(256) void k_fill_mx_den(float* mx, float* den, int n) {
    int i = blockIdx.x * 256 + threadIdx.x;
    if (i < n) { mx[i] = -3.0e38f; den[i] = 0.f; }
}

__device__ __forceinline__ void atomicMaxF(float* addr, float v) {
    // sign-aware integer trick (monotone under int-max for >=0, uint-min for <0)
    if (v >= 0.f) atomicMax((int*)addr, __float_as_int(v));
    else          atomicMin((unsigned int*)addr, __float_as_uint(v));
}

__global__ __launch_bounds__(256) void k_edge_max(const int* __restrict__ ei,
        const float* __restrict__ al_s, const float* __restrict__ al_d,
        float* __restrict__ mx, int E, int N) {
    int idx = blockIdx.x * 256 + threadIdx.x;
    int ET8 = (E + N) * HEADS;
    if (idx >= ET8) return;
    int e = idx >> 3, hh = idx & 7;
    int s, d;
    if (e < E) { s = ei[e]; d = ei[E + e]; } else { s = d = e - E; }
    float v = al_s[s * 8 + hh] + al_d[d * 8 + hh];
    v = v > 0.f ? v : 0.2f * v;                       // leaky_relu 0.2
    atomicMaxF(&mx[d * 8 + hh], v);
}

// one block per edge; un-normalized accumulate: acc[dst] += ex*h[src], den[dst] += ex
__global__ __launch_bounds__(256) void k_edge_accum(const int* __restrict__ ei,
        const float* __restrict__ al_s, const float* __restrict__ al_d,
        const float* __restrict__ mx, const float* __restrict__ h,
        float* __restrict__ den, float* __restrict__ acc, int E, int N) {
    int e = blockIdx.x;
    int t = threadIdx.x;
    int s, d;
    if (e < E) { s = ei[e]; d = ei[E + e]; } else { s = d = e - E; }
    const float* hs = h + (size_t)s * WCOLS;
    float* ap = acc + (size_t)d * WCOLS;
    #pragma unroll
    for (int hh = 0; hh < HEADS; ++hh) {
        float v = al_s[s * 8 + hh] + al_d[d * 8 + hh];
        v = v > 0.f ? v : 0.2f * v;
        float ex = __expf(v - mx[d * 8 + hh]);
        if (t == 0) atomicAdd(&den[d * 8 + hh], ex);
        atomicAdd(&ap[hh * 256 + t], ex * hs[hh * 256 + t]);
    }
}

// ---------------------------------------------------------------- finalize
// head-mean(acc/den) + bias [+ BN + ReLU -> bf16 next-layer input | f32 d_out]
__global__ __launch_bounds__(256) void k_finalize(const float* __restrict__ acc,
        const float* __restrict__ den, const float* __restrict__ bias,
        const float* __restrict__ g, const float* __restrict__ be,
        const float* __restrict__ bm, const float* __restrict__ bv,
        float* __restrict__ out_f32, __bf16* __restrict__ out_bf16, int do_bn) {
    int n = blockIdx.x, t = threadIdx.x;
    const float* ap = acc + (size_t)n * WCOLS;
    float val = 0.f;
    #pragma unroll
    for (int hh = 0; hh < HEADS; ++hh)
        val += ap[hh * 256 + t] / (den[n * 8 + hh] + 1e-16f);
    val = val * (1.0f / HEADS) + bias[t];
    if (do_bn) {
        val = (val - bm[t]) * rsqrtf(bv[t] + 1e-5f) * g[t] + be[t];
        val = fmaxf(val, 0.f);
        out_bf16[(size_t)n * 256 + t] = (__bf16)val;
    } else {
        out_f32[(size_t)n * 256 + t] = val;
    }
}

// ---------------------------------------------------------------- doc embed
__global__ __launch_bounds__(256) void k_doc(const float* __restrict__ doc,
        const float* __restrict__ W, const float* __restrict__ b,
        float* __restrict__ out) {
    int r = blockIdx.x, c = threadIdx.x;
    const float* dp = doc + (size_t)r * DOCIN;
    float a = b[c];
    for (int k = 0; k < DOCIN; ++k) a += dp[k] * W[(size_t)k * 256 + c];
    out[(size_t)r * 256 + c] = a;
}

// ---------------------------------------------------------------- launch
extern "C" void kernel_launch(void* const* d_in, const int* in_sizes, int n_in,
                              void* d_out, int out_size, void* d_ws, size_t ws_size,
                              hipStream_t stream) {
    const int N  = in_sizes[0] / HIDC;
    const int E  = in_sizes[1] / 2;
    const int ET = E + N;

    const float* x    = (const float*)d_in[0];
    const int*   ei   = (const int*)d_in[1];
    const float* doc  = (const float*)d_in[2];
    const float* W [3] = {(const float*)d_in[3], (const float*)d_in[7],  (const float*)d_in[11]};
    const float* As[3] = {(const float*)d_in[4], (const float*)d_in[8],  (const float*)d_in[12]};
    const float* Ad[3] = {(const float*)d_in[5], (const float*)d_in[9],  (const float*)d_in[13]};
    const float* Bi[3] = {(const float*)d_in[6], (const float*)d_in[10], (const float*)d_in[14]};
    const float* G [2] = {(const float*)d_in[15], (const float*)d_in[19]};
    const float* Be[2] = {(const float*)d_in[16], (const float*)d_in[20]};
    const float* Bm[2] = {(const float*)d_in[17], (const float*)d_in[21]};
    const float* Bv[2] = {(const float*)d_in[18], (const float*)d_in[22]};
    const float* docW  = (const float*)d_in[23];
    const float* docB  = (const float*)d_in[24];
    float* out = (float*)d_out;

    // workspace carve-up
    char* base = (char*)d_ws;
    size_t off = 0;
    auto carve = [&](size_t bytes) { char* p = base + off; off = (off + bytes + 255) & ~(size_t)255; return p; };
    float*  hbuf = (float*) carve((size_t)N * WCOLS * 4);   // GEMM out [N,8,256]
    float*  accb = (float*) carve((size_t)N * WCOLS * 4);   // edge accumulator
    __bf16* xb   = (__bf16*)carve((size_t)N * HIDC * 2);    // bf16 activations
    __bf16* wt   = (__bf16*)carve((size_t)WCOLS * HIDC * 2);// bf16 W^T
    float*  als  = (float*) carve((size_t)N * HEADS * 4);
    float*  ald  = (float*) carve((size_t)N * HEADS * 4);
    float*  mx   = (float*) carve((size_t)N * HEADS * 4);
    float*  den  = (float*) carve((size_t)N * HEADS * 4);
    (void)ws_size; (void)n_in; (void)out_size;

    const dim3 B256(256);
    k_f32_to_bf16<<<(int)(((size_t)N * HIDC + 255) / 256), B256, 0, stream>>>(x, xb, (size_t)N * HIDC);

    for (int l = 0; l < 3; ++l) {
        k_transpose_w<<<WCOLS * HIDC / 256, B256, 0, stream>>>(W[l], wt);
        k_gemm_wmma<<<dim3((N + 31) / 32, WCOLS / 256), B256, 0, stream>>>(xb, wt, hbuf, N);
        k_attn_logits<<<N, B256, 0, stream>>>(hbuf, As[l], Ad[l], als, ald, N);
        k_fill_mx_den<<<(N * HEADS + 255) / 256, B256, 0, stream>>>(mx, den, N * HEADS);
        hipMemsetAsync(accb, 0, (size_t)N * WCOLS * 4, stream);
        k_edge_max<<<(ET * HEADS + 255) / 256, B256, 0, stream>>>(ei, als, ald, mx, E, N);
        k_edge_accum<<<ET, B256, 0, stream>>>(ei, als, ald, mx, hbuf, den, accb, E, N);
        if (l < 2)
            k_finalize<<<N, B256, 0, stream>>>(accb, den, Bi[l], G[l], Be[l], Bm[l], Bv[l],
                                               nullptr, xb, 1);
        else
            k_finalize<<<N, B256, 0, stream>>>(accb, den, Bi[2], nullptr, nullptr, nullptr, nullptr,
                                               out, nullptr, 0);
    }
    k_doc<<<64, B256, 0, stream>>>(doc, docW, docB, out + (size_t)N * HIDC);
}
